// encoder_transformer_8624294330492
// MI455X (gfx1250) — matrix operand order repaired and checked by reading, not compiled, not run on hardware
//
#include <hip/hip_runtime.h>
#include <hip/hip_bf16.h>

typedef __bf16 bf16;
typedef float v8f   __attribute__((ext_vector_type(8)));
typedef bf16  v16bf __attribute__((ext_vector_type(16)));

union FragBF { v16bf v; uint4 u[2]; };

#define BM 128
#define BN 64
#define BK 64   // two 32-wide WMMA K-substeps per LDS stage

// CDNA5 async global->LDS DMA (per-lane 16B), tracked by ASYNCcnt.
__device__ __forceinline__ void async_ld_b128(const bf16* g, bf16* l) {
  const unsigned lds = (unsigned)(unsigned long long)l;     // addr[31:0] = LDS offset
  asm volatile("global_load_async_to_lds_b128 %0, %1, off"
               :: "v"(lds), "v"((unsigned long long)g)
               : "memory");
}
__device__ __forceinline__ void wait_async0() {
  asm volatile("s_wait_asynccnt 0" ::: "memory");
}

// C[M,N] = alpha * A[M,K] (bf16, row-major, lda) x Bt[N,K]^T (bf16, N-major, ldbt) + bias
// Optional f32 and/or bf16 outputs. gridDim.z batches with element strides.
__global__ __launch_bounds__(256) void gemm_bf16_wmma(
    const bf16* __restrict__ A, const bf16* __restrict__ Bt,
    const float* __restrict__ bias,
    float* __restrict__ outF, bf16* __restrict__ outB,
    int M, int N, int K, int lda, int ldbt, int ldc,
    size_t strideA, size_t strideB, size_t strideC, float alpha)
{
  __shared__ __align__(16) bf16 As[2][BM * BK];
  __shared__ __align__(16) bf16 Bs[2][BN * BK];

  const int tid  = threadIdx.x;
  const int lane = tid & 31;
  const int wave = tid >> 5;
  const int wm = wave >> 1;          // 0..3 : 32-row strip
  const int wn = wave & 1;           // 0..1 : 32-col strip
  const int hl = lane >> 4;          // lane half
  const int r  = lane & 15;
  const int bz = blockIdx.z;

  A  += (size_t)bz * strideA;
  Bt += (size_t)bz * strideB;
  const int m_blk = blockIdx.y * BM;
  const int n_blk = blockIdx.x * BN;

  v8f acc[2][2];
#pragma unroll
  for (int i = 0; i < 2; ++i)
#pragma unroll
    for (int j = 0; j < 2; ++j) {
      v8f z = {0.f, 0.f, 0.f, 0.f, 0.f, 0.f, 0.f, 0.f};
      acc[i][j] = z;
    }

  // Staging assignment: A tile row = 64 bf16 (128B) -> 2 threads x 64B;
  //                     B tile row = 64 bf16 (128B) -> 4 threads x 32B.
  const int arow = tid >> 1;          // 0..127
  const int aseg = (tid & 1) * 32;    // bf16 elems (0 / 32)
  const int brow = tid >> 2;          // 0..63
  const int bseg = (tid & 3) * 16;    // bf16 elems

  auto stage = [&](int k0, int buf) {
    const bf16* gA = A + (size_t)(m_blk + arow) * lda + k0 + aseg;
    bf16* sA = &As[buf][arow * BK + aseg];
#pragma unroll
    for (int c = 0; c < 4; ++c) async_ld_b128(gA + c * 8, sA + c * 8);
    const bf16* gB = Bt + (size_t)(n_blk + brow) * ldbt + k0 + bseg;
    bf16* sB = &Bs[buf][brow * BK + bseg];
#pragma unroll
    for (int c = 0; c < 2; ++c) async_ld_b128(gB + c * 8, sB + c * 8);
  };

  stage(0, 0);
  wait_async0();
  __syncthreads();

  int buf = 0;
  for (int k0 = 0; k0 < K; k0 += BK) {
    if (k0 + BK < K) stage(k0 + BK, buf ^ 1);   // async prefetch of next tile

#pragma unroll
    for (int ks = 0; ks < 2; ++ks) {            // two 32-wide K-substeps
      const int kb = ks * 32;
      FragBF fa[2], fb[2];
      // A frag: lane<16 -> K [0,8)+[16,24); lane>=16 -> K [8,16)+[24,32)
#pragma unroll
      for (int i = 0; i < 2; ++i) {
        const bf16* p = &As[buf][(wm * 32 + i * 16 + r) * BK + kb];
        fa[i].u[0] = *(const uint4*)(p + 8 * hl);
        fa[i].u[1] = *(const uint4*)(p + 16 + 8 * hl);
      }
      // B frag: column N=r, lane<16 -> K [0,8)+[16,24); lane>=16 -> K [8,16)+[24,32)
#pragma unroll
      for (int j = 0; j < 2; ++j) {
        const bf16* p = &Bs[buf][(wn * 32 + j * 16 + r) * BK + kb];
        fb[j].u[0] = *(const uint4*)(p + 8 * hl);
        fb[j].u[1] = *(const uint4*)(p + 16 + 8 * hl);
      }
#pragma unroll
      for (int i = 0; i < 2; ++i)
#pragma unroll
        for (int j = 0; j < 2; ++j)
          acc[i][j] = __builtin_amdgcn_wmma_f32_16x16x32_bf16(
              false, fa[i].v, false, fb[j].v, (short)0, acc[i][j], false, false);
    }

    wait_async0();       // next tile landed in LDS
    __syncthreads();     // all waves done reading current tile
    buf ^= 1;
  }

  // Epilogue: D layout VGPR j -> (M = 8*hl + j, N = r)
#pragma unroll
  for (int i = 0; i < 2; ++i) {
#pragma unroll
    for (int j = 0; j < 2; ++j) {
      const int col = n_blk + wn * 32 + j * 16 + r;
      const float bv = bias ? bias[col] : 0.f;
#pragma unroll
      for (int vv = 0; vv < 8; ++vv) {
        const int row = m_blk + wm * 32 + i * 16 + 8 * hl + vv;
        const float val = acc[i][j][vv] * alpha + bv;
        const size_t idx = (size_t)bz * strideC + (size_t)row * ldc + col;
        if (outF) outF[idx] = val;
        if (outB) outB[idx] = (bf16)val;
      }
    }
  }
}

// Softmax over the q axis (axis=1) of scores[b][q][k]; coalesced across k lanes.
__global__ __launch_bounds__(256) void softmax_q_axis(
    const float* __restrict__ scores, bf16* __restrict__ attn, int S)
{
  const int b = blockIdx.y;
  const int k = blockIdx.x * blockDim.x + threadIdx.x;
  const float* col = scores + (size_t)b * S * S + k;
  float mx = -3.4e38f;
  for (int q = 0; q < S; ++q) mx = fmaxf(mx, col[(size_t)q * S]);
  float sum = 0.f;
  for (int q = 0; q < S; ++q) sum += __expf(col[(size_t)q * S] - mx);
  const float inv = 1.f / sum;
  bf16* ocol = attn + (size_t)b * S * S + k;
  for (int q = 0; q < S; ++q)
    ocol[(size_t)q * S] = (bf16)(__expf(col[(size_t)q * S] - mx) * inv);
}

// out = LayerNorm(a + b) * g + beta ; writes f32 and bf16 copies. D=512, 256 thr x 2.
__global__ __launch_bounds__(256) void add_ln(
    const float* __restrict__ a, const float* __restrict__ b,
    const float* __restrict__ g, const float* __restrict__ beta,
    float* __restrict__ outF, bf16* __restrict__ outB, int D)
{
  const int row = blockIdx.x;
  const int tid = threadIdx.x;
  const size_t base = (size_t)row * D;
  const int c0 = tid * 2, c1 = tid * 2 + 1;
  const float x0 = a[base + c0] + b[base + c0];
  const float x1 = a[base + c1] + b[base + c1];
  __shared__ float rs[256], rq[256];
  rs[tid] = x0 + x1;
  rq[tid] = x0 * x0 + x1 * x1;
  __syncthreads();
  for (int s = 128; s > 0; s >>= 1) {
    if (tid < s) { rs[tid] += rs[tid + s]; rq[tid] += rq[tid + s]; }
    __syncthreads();
  }
  const float mean = rs[0] / D;
  const float var  = rq[0] / D - mean * mean;   // biased var (jnp.var)
  const float rstd = rsqrtf(var + 1e-5f);
  const float y0 = (x0 - mean) * rstd * g[c0] + beta[c0];
  const float y1 = (x1 - mean) * rstd * g[c1] + beta[c1];
  outF[base + c0] = y0; outF[base + c1] = y1;
  outB[base + c0] = (bf16)y0; outB[base + c1] = (bf16)y1;
}

__global__ void cvt_f32_bf16(const float* __restrict__ in, bf16* __restrict__ out, size_t n) {
  size_t i = (size_t)blockIdx.x * blockDim.x + threadIdx.x;
  if (i < n) out[i] = (bf16)in[i];
}

// wqkvT[n][k] (n<192, k<512), bf16, plus concatenated bias (f32).
__global__ void build_wqkvT(const float* wq, const float* wk, const float* wv,
                            const float* bq, const float* bk, const float* bv,
                            bf16* wqkvT, float* bqkv) {
  const int idx = blockIdx.x * 256 + threadIdx.x;
  if (idx >= 192 * 512) return;
  const int nn = idx / 512, kk = idx % 512;
  float v;
  if (nn < 64)       v = wq[kk * 64 + nn];
  else if (nn < 128) v = wk[kk * 64 + (nn - 64)];
  else               v = wv[kk * 64 + (nn - 128)];
  wqkvT[idx] = (bf16)v;
  if (kk == 0) bqkv[nn] = (nn < 64) ? bq[nn] : (nn < 128) ? bk[nn - 64] : bv[nn - 128];
}

// tile(head,8) @ wo  ==  head @ wo_folded ; woT[n][d] = sum_t wo[64t+d][n]
__global__ void build_woT(const float* wo, bf16* woT) {
  const int idx = blockIdx.x * 256 + threadIdx.x;
  if (idx >= 512 * 64) return;
  const int nn = idx / 64, dd = idx % 64;
  float s = 0.f;
  for (int t = 0; t < 8; ++t) s += wo[(size_t)(t * 64 + dd) * 512 + nn];
  woT[idx] = (bf16)s;
}

// wT[n*K+k] = w[k*N+n]  (bf16)
__global__ void build_wT(const float* w, bf16* wT, int K, int N) {
  const size_t idx = (size_t)blockIdx.x * 256 + threadIdx.x;
  if (idx >= (size_t)K * N) return;
  const int nn = (int)(idx / K), kk = (int)(idx % K);
  wT[idx] = (bf16)w[(size_t)kk * N + nn];
}

// vT[b][d][s] = qkv[(b*1024+s)*192 + 128 + d]
__global__ void vtrans(const bf16* __restrict__ qkv, bf16* __restrict__ vT) {
  const int idx = blockIdx.x * 256 + threadIdx.x;  // 16*64*1024
  const int b = idx >> 16;
  const int rem = idx & 65535;
  const int d = rem >> 10;
  const int s = rem & 1023;
  vT[idx] = qkv[((size_t)(b * 1024 + s)) * 192 + 128 + d];
}

extern "C" void kernel_launch(void* const* d_in, const int* in_sizes, int n_in,
                              void* d_out, int out_size, void* d_ws, size_t ws_size,
                              hipStream_t stream) {
  (void)in_sizes; (void)n_in; (void)out_size; (void)ws_size;
  const float* x   = (const float*)d_in[0];
  const float* wq  = (const float*)d_in[1];
  const float* bq  = (const float*)d_in[2];
  const float* wk  = (const float*)d_in[3];
  const float* bk  = (const float*)d_in[4];
  const float* wv  = (const float*)d_in[5];
  const float* bv  = (const float*)d_in[6];
  const float* wo  = (const float*)d_in[7];
  const float* bo  = (const float*)d_in[8];
  const float* lng = (const float*)d_in[9];
  const float* lnb = (const float*)d_in[10];
  const float* w1  = (const float*)d_in[11];
  const float* b1  = (const float*)d_in[12];
  const float* w2  = (const float*)d_in[13];
  const float* b2  = (const float*)d_in[14];

  constexpr int Bn = 16, S = 1024, D = 512, F = 2048, HD = 64;
  constexpr size_t M = (size_t)Bn * S;   // 16384 token rows

  char* ws = (char*)d_ws;
  size_t off = 0;
  auto alloc = [&](size_t bytes) -> char* {
    char* p = ws + off;
    off += (bytes + 255) & ~(size_t)255;
    return p;
  };
  bf16*  xb     = (bf16*) alloc(M * D * 2);
  float* x32    = (float*)alloc(M * D * 4);
  bf16*  qkv    = (bf16*) alloc(M * 192 * 2);
  bf16*  vT     = (bf16*) alloc((size_t)Bn * HD * S * 2);
  float* scores = (float*)alloc((size_t)Bn * S * S * 4);
  bf16*  attn   = (bf16*) alloc((size_t)Bn * S * S * 2);
  bf16*  head   = (bf16*) alloc(M * HD * 2);
  float* attno  = (float*)alloc(M * D * 4);
  float* h32    = (float*)alloc(M * D * 4);
  bf16*  hb     = (bf16*) alloc(M * D * 2);
  bf16*  ff1b   = (bf16*) alloc(M * F * 2);
  float* ff2    = (float*)alloc(M * D * 4);
  bf16*  wqkvT  = (bf16*) alloc((size_t)192 * D * 2);
  float* bqkv   = (float*)alloc(192 * 4);
  bf16*  woT    = (bf16*) alloc((size_t)D * HD * 2);
  bf16*  w1T    = (bf16*) alloc((size_t)F * D * 2);
  bf16*  w2T    = (bf16*) alloc((size_t)D * F * 2);

  // One-time prep (re-run every call: deterministic, graph-capture safe).
  cvt_f32_bf16<<<(unsigned)((M * D + 255) / 256), 256, 0, stream>>>(x, xb, M * D);
  build_wqkvT<<<(192 * 512 + 255) / 256, 256, 0, stream>>>(wq, wk, wv, bq, bk, bv, wqkvT, bqkv);
  build_woT<<<(512 * 64 + 255) / 256, 256, 0, stream>>>(wo, woT);
  build_wT<<<(unsigned)(((size_t)D * F + 255) / 256), 256, 0, stream>>>(w1, w1T, D, F);
  build_wT<<<(unsigned)(((size_t)F * D + 255) / 256), 256, 0, stream>>>(w2, w2T, F, D);

  const float* xf_in = x;
  for (int layer = 0; layer < 4; ++layer) {
    // QKV: [M,512] x [512,192] -> qkv bf16 (ldc=192)
    gemm_bf16_wmma<<<dim3(192 / BN, (unsigned)(M / BM), 1), 256, 0, stream>>>(
        xb, wqkvT, bqkv, nullptr, qkv,
        (int)M, 192, D, D, D, 192, 0, 0, 0, 1.f);
    vtrans<<<(Bn * HD * S) / 256, 256, 0, stream>>>(qkv, vT);
    // scores = (q @ k^T)/8 : per-batch, k row-major is exactly Bt
    gemm_bf16_wmma<<<dim3(S / BN, S / BM, Bn), 256, 0, stream>>>(
        qkv, qkv + 64, nullptr, scores, nullptr,
        S, S, HD, 192, 192, S,
        (size_t)S * 192, (size_t)S * 192, (size_t)S * S, 0.125f);
    softmax_q_axis<<<dim3(S / 256, Bn), 256, 0, stream>>>(scores, attn, S);
    // head = attn @ v : Bt = vT
    gemm_bf16_wmma<<<dim3(HD / BN, S / BM, Bn), 256, 0, stream>>>(
        attn, vT, nullptr, nullptr, head,
        S, HD, S, S, S, HD,
        (size_t)S * S, (size_t)HD * S, (size_t)S * HD, 1.f);
    // attn_out = head @ wo_folded + bo  (concat fold: K=64 instead of 512)
    gemm_bf16_wmma<<<dim3(D / BN, (unsigned)(M / BM), 1), 256, 0, stream>>>(
        head, woT, bo, attno, nullptr,
        (int)M, D, HD, HD, HD, D, 0, 0, 0, 1.f);
    // h = LN(attn_out + x_in)
    add_ln<<<(unsigned)M, 256, 0, stream>>>(attno, xf_in, lng, lnb, h32, hb, D);
    // ff1 = h @ w1 + b1 (bf16 out; no ReLU per reference)
    gemm_bf16_wmma<<<dim3(F / BN, (unsigned)(M / BM), 1), 256, 0, stream>>>(
        hb, w1T, b1, nullptr, ff1b,
        (int)M, F, D, D, D, F, 0, 0, 0, 1.f);
    // ff2 = ff1 @ w2 + b2 (f32 out)
    gemm_bf16_wmma<<<dim3(D / BN, (unsigned)(M / BM), 1), 256, 0, stream>>>(
        ff1b, w2T, b2, ff2, nullptr,
        (int)M, D, F, F, F, D, 0, 0, 0, 1.f);
    // x = LN(h + ff) ; last layer writes f32 straight into d_out
    float* xoutF = (layer == 3) ? (float*)d_out : x32;
    add_ln<<<(unsigned)M, 256, 0, stream>>>(h32, ff2, lng, lnb, xoutF, xb, D);
    xf_in = x32;
  }
}
